// AttentionThinkingBlock_1005022347436
// MI455X (gfx1250) — matrix operand order repaired and checked by reading, not compiled, not run on hardware
//
#include <hip/hip_runtime.h>
#include <hip/hip_bf16.h>

typedef __bf16 bf16_t;
typedef __attribute__((ext_vector_type(16))) __bf16 v16bf;
typedef __attribute__((ext_vector_type(8)))  __bf16 v8bf;
typedef __attribute__((ext_vector_type(8)))  float  v8f;
typedef unsigned int u32;
typedef __attribute__((ext_vector_type(4))) unsigned int u32x4;
typedef __attribute__((ext_vector_type(8))) int i32x8;
typedef __attribute__((ext_vector_type(4))) int i32x4;

#define BT  2
#define SEQ 2048
#define DIM 1024
#define NH  16
#define HS  64

#if defined(__AMDGCN__) && __has_builtin(__builtin_amdgcn_tensor_load_to_lds) && \
    __has_builtin(__builtin_amdgcn_s_wait_tensorcnt)
#define HAS_TDM 1
#else
#define HAS_TDM 0
#endif

// ---------------- WMMA fragment helpers (CDNA5 16x16x32 bf16 layouts) --------
// A (16x32, MxK): lane = m + 16*h. Per lane: K = 8h..8h+7 and 16+8h..16+8h+7.
__device__ inline v16bf load_a_frag(const bf16_t* A, int lda, int m, int h, int k0) {
  const bf16_t* p = A + (size_t)m * lda + k0 + 8 * h;
  v8bf lo = *(const v8bf*)(p);
  v8bf hi = *(const v8bf*)(p + 16);
  v16bf r;
#pragma unroll
  for (int i = 0; i < 8; ++i) { r[i] = lo[i]; r[8 + i] = hi[i]; }
  return r;
}

// B (32x16, KxN) sourced from a row-major [N,K] ("transposed") matrix.
// lane = n + 16*h. Per lane: K = 8h..8h+7 and 16+8h..16+8h+7.
__device__ inline v16bf load_b_frag(const bf16_t* Bt, int ldb, int n, int h, int k0) {
  const bf16_t* p = Bt + (size_t)n * ldb + k0 + 8 * h;
  v8bf lo = *(const v8bf*)(p);
  v8bf hi = *(const v8bf*)(p + 16);
  v16bf r;
#pragma unroll
  for (int i = 0; i < 8; ++i) { r[i] = lo[i]; r[8 + i] = hi[i]; }
  return r;
}

__device__ inline v8f wmma_bf16(v16bf a, v16bf b, v8f c) {
  return __builtin_amdgcn_wmma_f32_16x16x32_bf16(false, a, false, b, (short)0, c,
                                                 false, false);
}

// ---------------- B-panel staging: TDM (tensor_load_to_lds) or fallback ------
// Stage a [256 rows x 32 K] bf16 panel of Bt into LDS (row stride 32).
__device__ inline void stage_b_tile(bf16_t* lds, const bf16_t* Bt, int ldb,
                                    int n0, int k0, int wave) {
#if HAS_TDM
  if (wave == 0) {
    const unsigned long long ga =
        (unsigned long long)(size_t)(Bt + (size_t)n0 * ldb + k0);
    u32x4 g0;
    g0[0] = 1u;                                    // count=1 valid descriptor
    g0[1] = (u32)(size_t)lds;                      // lds_addr (flat[31:0] = LDS offset)
    g0[2] = (u32)ga;                               // global_addr[31:0]
    g0[3] = ((u32)(ga >> 32) & 0x1FFFFFFu) | (2u << 30);  // addr[56:32] | type=2
    i32x8 g1;
    g1[0] = (int)(1u << 16);      // workgroup_mask=0, data_size=1 (2 bytes)
    g1[1] = (int)(32u << 16);     // tensor_dim0[15:0]=32 at bits 63:48
    g1[2] = (int)(256u << 16);    // tensor_dim0 hi=0 | tensor_dim1[15:0]=256
    g1[3] = (int)(32u << 16);     // tensor_dim1 hi=0 | tile_dim0=32
    g1[4] = 256;                  // tile_dim1=256, tile_dim2=0
    g1[5] = ldb;                  // tensor_dim0_stride[31:0] (elements)
    g1[6] = 0;                    // stride hi, dim1_stride lo
    g1[7] = 0;
    i32x4 z4 = {};
#if __clang_major__ >= 23
    i32x8 z8 = {};
    __builtin_amdgcn_tensor_load_to_lds(g0, g1, z4, z4, z8, 0);
#else
    __builtin_amdgcn_tensor_load_to_lds(g0, g1, z4, z4, 0);
#endif
  }
#else
  const int t = threadIdx.x;  // 256 threads <-> 256 rows
  const bf16_t* src = Bt + (size_t)(n0 + t) * ldb + k0;
  v16bf c0 = *(const v16bf*)src;
  v16bf c1 = *(const v16bf*)(src + 16);
  *(v16bf*)(lds + t * 32) = c0;
  *(v16bf*)(lds + t * 32 + 16) = c1;
#endif
}

__device__ inline void stage_commit(int wave) {
#if HAS_TDM
  if (wave == 0) __builtin_amdgcn_s_wait_tensorcnt((short)0);
#endif
  __syncthreads();
}

// ---------------- generic epilogue-flavored GEMM -----------------------------
// C = act(A[MxK] * B[KxN] + bias (+ res)).  Bt is [N,K] row-major bf16.
// Block = 8 waves (2 M x 4 N); wave computes 32x64; block tile 64x256.
// B panel (256x32) staged in LDS via TDM, double-buffered.
template <bool BIAS, bool RELU, bool RES, bool OUTF32>
__global__ void gemm_wmma_kernel(const bf16_t* __restrict__ A, int lda,
                                 const bf16_t* __restrict__ Bt, int ldb,
                                 const float* __restrict__ bias,
                                 const float* __restrict__ res, int ldr,
                                 float* __restrict__ C32,
                                 bf16_t* __restrict__ Cb, int ldc, int K) {
  __shared__ bf16_t bstage[2][256 * 32];
  const int lane = threadIdx.x & 31;
  const int wave = threadIdx.x >> 5;
  const int wx = wave & 3, wy = wave >> 2;
  const int M0 = blockIdx.y * 64 + wy * 32;
  const int N0 = blockIdx.x * 256;
  const int m = lane & 15, h = lane >> 4;

  v8f acc0[4], acc1[4];
#pragma unroll
  for (int j = 0; j < 4; ++j) { acc0[j] = (v8f){}; acc1[j] = (v8f){}; }

  stage_b_tile(bstage[0], Bt, ldb, N0, 0, wave);
  stage_commit(wave);

  for (int k0 = 0; k0 < K; k0 += 32) {
    const int p = (k0 >> 5) & 1;
    if (k0 + 32 < K) {
      stage_b_tile(bstage[p ^ 1], Bt, ldb, N0, k0 + 32, wave);  // overlap DMA
      __builtin_prefetch(A + (size_t)(M0 + m) * lda + k0 + 32, 0, 0);
    }
    v16bf a0 = load_a_frag(A, lda, M0 + m, h, k0);
    v16bf a1 = load_a_frag(A, lda, M0 + 16 + m, h, k0);
#pragma unroll
    for (int j = 0; j < 4; ++j) {
      v16bf b = load_b_frag(bstage[p], 32, wx * 64 + 16 * j + m, h, 0);
      acc0[j] = wmma_bf16(a0, b, acc0[j]);
      acc1[j] = wmma_bf16(a1, b, acc1[j]);
    }
    stage_commit(wave);
  }

#pragma unroll
  for (int j = 0; j < 4; ++j) {
    const int n = N0 + wx * 64 + 16 * j + m;
    float bv = 0.f;
    if (BIAS) bv = bias[n];
#pragma unroll
    for (int r = 0; r < 8; ++r) {
      {
        const int mm = M0 + r + 8 * h;
        float v = acc0[j][r] + bv;
        if (RES) v += res[(size_t)mm * ldr + n];
        if (RELU) v = fmaxf(v, 0.f);
        if (OUTF32) C32[(size_t)mm * ldc + n] = v;
        else        Cb[(size_t)mm * ldc + n] = (bf16_t)v;
      }
      {
        const int mm = M0 + 16 + r + 8 * h;
        float v = acc1[j][r] + bv;
        if (RES) v += res[(size_t)mm * ldr + n];
        if (RELU) v = fmaxf(v, 0.f);
        if (OUTF32) C32[(size_t)mm * ldc + n] = v;
        else        Cb[(size_t)mm * ldc + n] = (bf16_t)v;
      }
    }
  }
}

// ---------------- fused QKV projection (wave = 16x64, full head width) -------
// y encodes (mat, b, head). V is stored transposed [B,H,HS,T] for attention.
__global__ void qkv_wmma_kernel(const bf16_t* __restrict__ xn,
                                const bf16_t* __restrict__ wqT,
                                const bf16_t* __restrict__ wkT,
                                const bf16_t* __restrict__ wvT,
                                bf16_t* __restrict__ qo, bf16_t* __restrict__ ko,
                                bf16_t* __restrict__ vto) {
  const int z = blockIdx.y;
  const int mat = z / (BT * NH);
  const int bh = z % (BT * NH);
  const int hd = bh % NH;
  const int lane = threadIdx.x & 31;
  const int wave = threadIdx.x >> 5;
  const int m0 = (blockIdx.x * 8 + wave) * 16;  // t tile
  const int m = lane & 15, h = lane >> 4;

  const bf16_t* A = xn + (size_t)(bh / NH) * SEQ * DIM;
  const bf16_t* W = (mat == 0 ? wqT : mat == 1 ? wkT : wvT) + (size_t)hd * HS * DIM;

  v8f acc[4];
#pragma unroll
  for (int j = 0; j < 4; ++j) acc[j] = (v8f){};
  for (int k0 = 0; k0 < DIM; k0 += 32) {
    v16bf a = load_a_frag(A, DIM, m0 + m, h, k0);
#pragma unroll
    for (int j = 0; j < 4; ++j) {
      v16bf bb = load_b_frag(W, DIM, 16 * j + m, h, k0);
      acc[j] = wmma_bf16(a, bb, acc[j]);
    }
  }
  if (mat == 2) {
    bf16_t* outp = vto + (size_t)bh * HS * SEQ;  // [hs][t]
#pragma unroll
    for (int j = 0; j < 4; ++j)
#pragma unroll
      for (int r = 0; r < 8; ++r)
        outp[(size_t)(16 * j + m) * SEQ + m0 + r + 8 * h] = (bf16_t)acc[j][r];
  } else {
    bf16_t* outp = (mat == 0 ? qo : ko) + (size_t)bh * SEQ * HS;  // [t][hs]
#pragma unroll
    for (int j = 0; j < 4; ++j)
#pragma unroll
      for (int r = 0; r < 8; ++r)
        outp[(size_t)(m0 + r + 8 * h) * HS + 16 * j + m] = (bf16_t)acc[j][r];
  }
}

// ---------------- causal flash attention (one wave = 16 queries) -------------
__global__ void attn_kernel(const bf16_t* __restrict__ q,   // [B,H,T,HS]
                            const bf16_t* __restrict__ k,   // [B,H,T,HS]
                            const bf16_t* __restrict__ vt,  // [B,H,HS,T]
                            bf16_t* __restrict__ ao) {      // [B,T,D]
  __shared__ bf16_t pbuf[4][16 * 32];
  const int wave = threadIdx.x >> 5;
  const int lane = threadIdx.x & 31;
  const int m = lane & 15, h = lane >> 4;
  const int bh = blockIdx.y;
  const int b = bh / NH, hd = bh % NH;
  const int q0 = blockIdx.x * 64 + wave * 16;

  const bf16_t* qp = q + ((size_t)bh * SEQ + q0) * HS;
  const bf16_t* kp = k + (size_t)bh * SEQ * HS;
  const bf16_t* vp = vt + (size_t)bh * HS * SEQ;
  bf16_t* lp = pbuf[wave];

  const v16bf aQ0 = load_a_frag(qp, HS, m, h, 0);
  const v16bf aQ1 = load_a_frag(qp, HS, m, h, 32);

  float mrow[8], lrow[8], alpha[8], pA[8], pB[8];
  v8f o0 = {}, o1 = {}, o2 = {}, o3 = {};
#pragma unroll
  for (int r = 0; r < 8; ++r) { mrow[r] = -1e30f; lrow[r] = 0.f; }
  const float scale = 0.125f;  // 1/sqrt(HS)

  for (int s0 = 0; s0 <= q0 + 15; s0 += 32) {
    // S = Q K^T for 32 keys. K[t][hs] is already [N,K] layout for B fragments.
    v8f sA = {}, sB = {};
    sA = wmma_bf16(aQ0, load_b_frag(kp + (size_t)s0 * HS, HS, m, h, 0), sA);
    sA = wmma_bf16(aQ1, load_b_frag(kp + (size_t)s0 * HS, HS, m, h, 32), sA);
    sB = wmma_bf16(aQ0, load_b_frag(kp + (size_t)(s0 + 16) * HS, HS, m, h, 0), sB);
    sB = wmma_bf16(aQ1, load_b_frag(kp + (size_t)(s0 + 16) * HS, HS, m, h, 32), sB);

#pragma unroll
    for (int r = 0; r < 8; ++r) {
      const int qi = q0 + r + 8 * h;
      float a = sA[r] * scale;
      float bb = sB[r] * scale;
      if (s0 + m > qi)      a = -1e30f;   // causal mask
      if (s0 + 16 + m > qi) bb = -1e30f;
      float t = fmaxf(a, bb);
#pragma unroll
      for (int off = 1; off < 16; off <<= 1) t = fmaxf(t, __shfl_xor(t, off, 32));
      const float mn = fmaxf(mrow[r], t);
      alpha[r] = __expf(mrow[r] - mn);
      const float ea = __expf(a - mn);
      const float eb = __expf(bb - mn);
      pA[r] = ea; pB[r] = eb;
      float rs = ea + eb;
#pragma unroll
      for (int off = 1; off < 16; off <<= 1) rs += __shfl_xor(rs, off, 32);
      lrow[r] = lrow[r] * alpha[r] + rs;
      mrow[r] = mn;
    }
#pragma unroll
    for (int r = 0; r < 8; ++r) {
      o0[r] *= alpha[r]; o1[r] *= alpha[r]; o2[r] *= alpha[r]; o3[r] *= alpha[r];
    }
    // Reshape P: score-acc layout -> A-fragment layout via per-wave LDS slab.
#pragma unroll
    for (int r = 0; r < 8; ++r) {
      const int row = r + 8 * h;
      lp[row * 32 + m]      = (bf16_t)pA[r];
      lp[row * 32 + 16 + m] = (bf16_t)pB[r];
    }
    asm volatile("" ::: "memory");  // same-wave DS ops are in-order; keep compiler honest
    const v16bf aP = load_a_frag(lp, 32, m, h, 0);
    asm volatile("" ::: "memory");
    // O += P V ; vt is [hs][t] => [N,K] layout for B fragments (K=key pos).
    o0 = wmma_bf16(aP, load_b_frag(vp + s0, SEQ, m,      h, 0), o0);
    o1 = wmma_bf16(aP, load_b_frag(vp + s0, SEQ, 16 + m, h, 0), o1);
    o2 = wmma_bf16(aP, load_b_frag(vp + s0, SEQ, 32 + m, h, 0), o2);
    o3 = wmma_bf16(aP, load_b_frag(vp + s0, SEQ, 48 + m, h, 0), o3);
  }

#pragma unroll
  for (int r = 0; r < 8; ++r) {
    const float inv = 1.f / lrow[r];
    const int t = q0 + r + 8 * h;
    bf16_t* orow = ao + ((size_t)b * SEQ + t) * DIM + hd * HS + m;
    orow[0]  = (bf16_t)(o0[r] * inv);
    orow[16] = (bf16_t)(o1[r] * inv);
    orow[32] = (bf16_t)(o2[r] * inv);
    orow[48] = (bf16_t)(o3[r] * inv);
  }
}

// ---------------- LayerNorm (f32 in, bf16 out), one block per row ------------
__global__ void ln_kernel(const float* __restrict__ x, const float* __restrict__ g,
                          const float* __restrict__ be, bf16_t* __restrict__ outp) {
  const int row = blockIdx.x;
  const float* xr = x + (size_t)row * DIM;
  float v[4], s = 0.f, ss = 0.f;
#pragma unroll
  for (int i = 0; i < 4; ++i) {
    v[i] = xr[threadIdx.x + i * 256];
    s += v[i]; ss += v[i] * v[i];
  }
#pragma unroll
  for (int off = 1; off < 32; off <<= 1) {
    s += __shfl_xor(s, off, 32);
    ss += __shfl_xor(ss, off, 32);
  }
  __shared__ float red[2][8];
  const int wave = threadIdx.x >> 5, lane = threadIdx.x & 31;
  if (lane == 0) { red[0][wave] = s; red[1][wave] = ss; }
  __syncthreads();
  s = 0.f; ss = 0.f;
#pragma unroll
  for (int w = 0; w < 8; ++w) { s += red[0][w]; ss += red[1][w]; }
  const float mu = s * (1.f / DIM);
  const float var = ss * (1.f / DIM) - mu * mu;
  const float rinv = rsqrtf(var + 1e-5f);
  bf16_t* orow = outp + (size_t)row * DIM;
#pragma unroll
  for (int i = 0; i < 4; ++i) {
    const int c = threadIdx.x + i * 256;
    orow[c] = (bf16_t)((v[i] - mu) * rinv * g[c] + be[c]);
  }
}

// ---------------- weight downconvert + transpose f32[K,N] -> bf16[N,K] -------
__global__ void transpose_bf16_kernel(const float* __restrict__ in,
                                      bf16_t* __restrict__ outp, int K, int N) {
  const size_t total = (size_t)K * N;
  const size_t idx = (size_t)blockIdx.x * blockDim.x + threadIdx.x;
  const size_t boff = (size_t)blockIdx.z * total;
  if (idx < total) {
    const int kk = (int)(idx / N), n = (int)(idx % N);
    outp[boff + (size_t)n * K + kk] = (bf16_t)in[boff + idx];
  }
}

// -----------------------------------------------------------------------------
extern "C" void kernel_launch(void* const* d_in, const int* in_sizes, int n_in,
                              void* d_out, int out_size, void* d_ws, size_t ws_size,
                              hipStream_t stream) {
  (void)in_sizes; (void)n_in; (void)out_size; (void)ws_size;
  const float* x     = (const float*)d_in[0];
  const float* wq    = (const float*)d_in[1];
  const float* wk    = (const float*)d_in[2];
  const float* wv    = (const float*)d_in[3];
  const float* wproj = (const float*)d_in[4];
  const float* bproj = (const float*)d_in[5];
  const float* w1    = (const float*)d_in[6];
  const float* b1    = (const float*)d_in[7];
  const float* w2    = (const float*)d_in[8];
  const float* b2    = (const float*)d_in[9];
  const float* ln1g  = (const float*)d_in[10];
  const float* ln1b  = (const float*)d_in[11];
  const float* ln2g  = (const float*)d_in[12];
  const float* ln2b  = (const float*)d_in[13];
  float* outp = (float*)d_out;

  char* ws = (char*)d_ws;
  const size_t MB = 1024 * 1024;
  bf16_t* xn  = (bf16_t*)(ws + 0 * MB);   // [4096,1024]        8MB
  bf16_t* hn  = (bf16_t*)(ws + 8 * MB);   // [4096,1024]        8MB
  bf16_t* qb  = (bf16_t*)(ws + 16 * MB);  // [B,H,T,HS]         8MB
  bf16_t* kb  = (bf16_t*)(ws + 24 * MB);  // [B,H,T,HS]         8MB
  bf16_t* vtb = (bf16_t*)(ws + 32 * MB);  // [B,H,HS,T]         8MB
  bf16_t* ao  = (bf16_t*)(ws + 40 * MB);  // [B,T,D]            8MB
  bf16_t* ff1 = (bf16_t*)(ws + 16 * MB);  // [4096,4096] 32MB (reuses q/k/vt/ao)
  bf16_t* wqT = (bf16_t*)(ws + 48 * MB);  // [H,HS,D]           2MB
  bf16_t* wkT = (bf16_t*)(ws + 50 * MB);
  bf16_t* wvT = (bf16_t*)(ws + 52 * MB);
  bf16_t* wpT = (bf16_t*)(ws + 54 * MB);  // [D,D]              2MB
  bf16_t* w1T = (bf16_t*)(ws + 56 * MB);  // [4D,D]             8MB
  bf16_t* w2T = (bf16_t*)(ws + 64 * MB);  // [D,4D]             8MB

  // Weight downconvert/transpose
  transpose_bf16_kernel<<<dim3((DIM * HS + 255) / 256, 1, NH), 256, 0, stream>>>(wq, wqT, DIM, HS);
  transpose_bf16_kernel<<<dim3((DIM * HS + 255) / 256, 1, NH), 256, 0, stream>>>(wk, wkT, DIM, HS);
  transpose_bf16_kernel<<<dim3((DIM * HS + 255) / 256, 1, NH), 256, 0, stream>>>(wv, wvT, DIM, HS);
  transpose_bf16_kernel<<<dim3((DIM * DIM + 255) / 256, 1, 1), 256, 0, stream>>>(wproj, wpT, DIM, DIM);
  transpose_bf16_kernel<<<dim3((DIM * 4 * DIM + 255) / 256, 1, 1), 256, 0, stream>>>(w1, w1T, DIM, 4 * DIM);
  transpose_bf16_kernel<<<dim3((4 * DIM * DIM + 255) / 256, 1, 1), 256, 0, stream>>>(w2, w2T, 4 * DIM, DIM);

  // LN1
  ln_kernel<<<BT * SEQ, 256, 0, stream>>>(x, ln1g, ln1b, xn);

  // QKV (V written transposed)
  qkv_wmma_kernel<<<dim3(SEQ / 128, 3 * BT * NH), 256, 0, stream>>>(
      xn, wqT, wkT, wvT, qb, kb, vtb);

  // Causal attention
  attn_kernel<<<dim3(SEQ / 64, BT * NH), 128, 0, stream>>>(qb, kb, vtb, ao);

  // h = x + ao @ wproj + bproj   (f32 into d_out)
  gemm_wmma_kernel<true, false, true, true>
      <<<dim3(DIM / 256, (BT * SEQ) / 64), 256, 0, stream>>>(
          ao, DIM, wpT, DIM, bproj, x, DIM, outp, nullptr, DIM, DIM);

  // LN2 on h
  ln_kernel<<<BT * SEQ, 256, 0, stream>>>(outp, ln2g, ln2b, hn);

  // ff1 = relu(hn @ w1 + b1)  (bf16)
  gemm_wmma_kernel<true, true, false, false>
      <<<dim3(4 * DIM / 256, (BT * SEQ) / 64), 256, 0, stream>>>(
          hn, DIM, w1T, DIM, b1, nullptr, 0, nullptr, ff1, 4 * DIM, DIM);

  // out = h + ff1 @ w2 + b2  (reads h from d_out, writes d_out 1:1 per thread)
  gemm_wmma_kernel<true, false, true, true>
      <<<dim3(DIM / 256, (BT * SEQ) / 64), 256, 0, stream>>>(
          ff1, 4 * DIM, w2T, 4 * DIM, b2, outp, DIM, outp, nullptr, DIM, 4 * DIM);
}
